// Selector_59777354826202
// MI455X (gfx1250) — matrix operand order repaired and checked by reading, not compiled, not run on hardware
//
#include <hip/hip_runtime.h>
#include <hip/hip_bf16.h>
#include <math.h>

#define BB   16
#define SS   64
#define CLn  400
#define QLn  64
#define DD   512
#define CQDn 512
#define LATn 256
#define VV   32000
#define K3   1536   // 3*D
#define K4   2048   // 3*D + CQD
#define ROWS 1024   // B*S

typedef __attribute__((ext_vector_type(16))) __bf16 v16bf;
typedef __attribute__((ext_vector_type(8)))  __bf16 v8bf;
typedef __attribute__((ext_vector_type(8)))  float  v8f;

// ---------------------------------------------------------------------------
// 1) Pack ctx3 = [ctx_q ; ctx_c ; decoded] into bf16 A [ROWS x K3]
// ---------------------------------------------------------------------------
__global__ __launch_bounds__(256) void build_A_bf16(
    const float* __restrict__ ctx_q, const float* __restrict__ ctx_c,
    const float* __restrict__ decoded, unsigned short* __restrict__ A)
{
  int idx = blockIdx.x * 256 + threadIdx.x;
  if (idx >= ROWS * K3) return;
  int row = idx / K3;
  int k   = idx - row * K3;
  float x;
  if (k < DD)            x = ctx_q  [(size_t)row * DD + k];
  else if (k < 2 * DD)   x = ctx_c  [(size_t)row * DD + (k - DD)];
  else                   x = decoded[(size_t)row * DD + (k - 2 * DD)];
  ((__bf16*)A)[idx] = (__bf16)x;
}

// ---------------------------------------------------------------------------
// 2) Transpose + convert W_vocab [K3 x VV] f32 -> WT [VV x K3] bf16 (LDS tiled)
// ---------------------------------------------------------------------------
__global__ __launch_bounds__(256) void transpose_W_bf16(
    const float* __restrict__ W, unsigned short* __restrict__ WT)
{
  __shared__ float tile[32][33];
  int vbase = blockIdx.x * 32;
  int kbase = blockIdx.y * 32;
  int tx = threadIdx.x;   // 0..31
  int ty = threadIdx.y;   // 0..7
  #pragma unroll
  for (int i = 0; i < 32; i += 8)
    tile[ty + i][tx] = W[(size_t)(kbase + ty + i) * VV + (vbase + tx)];
  __syncthreads();
  __bf16* wt = (__bf16*)WT;
  #pragma unroll
  for (int i = 0; i < 32; i += 8)
    wt[(size_t)(vbase + ty + i) * K3 + (kbase + tx)] = (__bf16)tile[tx][ty + i];
}

// ---------------------------------------------------------------------------
// 3) Gumbel-softmax estimator: est[row][3]
// ---------------------------------------------------------------------------
__global__ __launch_bounds__(256) void est_kernel(
    const float* __restrict__ ctx_q, const float* __restrict__ ctx_c,
    const float* __restrict__ decoded, const float* __restrict__ cq,
    const float* __restrict__ eps, const float* __restrict__ gumbel,
    const float* __restrict__ Wmu, const float* __restrict__ Wsg,
    const float* __restrict__ Wpi, const float* __restrict__ bpi,
    const int* __restrict__ num_step, float* __restrict__ est)
{
  __shared__ float ctx4[K4];
  __shared__ float h[LATn];
  __shared__ float ps[3];
  int row = blockIdx.x;
  int tid = threadIdx.x;
  for (int i = tid; i < DD; i += 256) {
    ctx4[i]          = ctx_q  [(size_t)row * DD + i];
    ctx4[DD + i]     = ctx_c  [(size_t)row * DD + i];
    ctx4[2 * DD + i] = decoded[(size_t)row * DD + i];
    ctx4[3 * DD + i] = cq     [(size_t)row * DD + i];
  }
  __syncthreads();
  float am = 0.f, as = 0.f;
  for (int k = 0; k < K4; ++k) {
    float x = ctx4[k];
    am = fmaf(x, Wmu[(size_t)k * LATn + tid], am);
    as = fmaf(x, Wsg[(size_t)k * LATn + tid], as);
  }
  float mu = tanhf(am);
  float sg = __expf(tanhf(as));
  h[tid] = mu + sg * eps[(size_t)row * LATn + tid];
  __syncthreads();
  if (tid < 3) {
    float p = bpi[tid];
    for (int j = 0; j < LATn; ++j) p = fmaf(h[j], Wpi[j * 3 + tid], p);
    ps[tid] = p;
  }
  __syncthreads();
  if (tid == 0) {
    float p0 = ps[0], p1 = ps[1], p2 = ps[2];
    float mx = fmaxf(p0, fmaxf(p1, p2));
    float lse = mx + logf(__expf(p0 - mx) + __expf(p1 - mx) + __expf(p2 - mx));
    float tau = 1.0f / (float)(*num_step);           // TEMP(=1.0) / num_step
    float z0 = (gumbel[row * 3 + 0] + (p0 - lse)) / tau;
    float z1 = (gumbel[row * 3 + 1] + (p1 - lse)) / tau;
    float z2 = (gumbel[row * 3 + 2] + (p2 - lse)) / tau;
    float zm = fmaxf(z0, fmaxf(z1, z2));
    float e0 = __expf(z0 - zm), e1 = __expf(z1 - zm), e2 = __expf(z2 - zm);
    float inv = 1.0f / (e0 + e1 + e2);
    est[row * 3 + 0] = e0 * inv;
    est[row * 3 + 1] = e1 * inv;
    est[row * 3 + 2] = e2 * inv;
  }
}

// ---------------------------------------------------------------------------
// 4) WMMA bf16 GEMM: logits[ROWS x VV] = A[ROWS x K3] * WT^T + bias
//    one wave computes a 32(M) x 64(N) tile: 2 A-frags x 4 B-frags ->
//    8 accumulators, 24 loaded bytes per WMMA (halves L2 re-read of WT
//    vs a 16xN tile). 8 waves / block.
// ---------------------------------------------------------------------------
__global__ __launch_bounds__(256) void gemm_logits(
    const unsigned short* __restrict__ Apk, const unsigned short* __restrict__ WTpk,
    const float* __restrict__ bias, float* __restrict__ out)
{
  const __bf16* A  = (const __bf16*)Apk;
  const __bf16* WT = (const __bf16*)WTpk;
  int lane = threadIdx.x & 31;
  int w = blockIdx.x * 8 + (threadIdx.x >> 5);   // 0..15999
  int mtile = w / 500;                           // 0..31  (M in steps of 32)
  int ntile = w - mtile * 500;                   // 0..499 (N in steps of 64)
  int m0 = mtile * 32;
  int n0 = ntile * 64;
  int half = lane >> 4;
  int l16  = lane & 15;

  // A-operand k order used here (contiguous, same as B): lane<16 -> K in [0..15];
  // lane>=16 -> K in [16..31]; row = lane%16
  const __bf16* ga0 = A + (size_t)(m0 + l16) * K3;
  const __bf16* ga1 = A + (size_t)(m0 + 16 + l16) * K3;
  int koff = half * 16;

  // B-operand k order used here (contiguous, same as A): lanes 0-15 hold K=k0..k0+15 (col=lane),
  // lanes 16-31 hold K=k0+16..k0+31 -> contiguous 32B per lane from WT[n][.]
  const __bf16* gb[4];
  #pragma unroll
  for (int t = 0; t < 4; ++t)
    gb[t] = WT + (size_t)(n0 + t * 16 + l16) * K3 + half * 16;

  v8f acc[2][4];
  #pragma unroll
  for (int i = 0; i < 2; ++i)
    #pragma unroll
    for (int t = 0; t < 4; ++t)
      acc[i][t] = (v8f){0.f,0.f,0.f,0.f,0.f,0.f,0.f,0.f};

  for (int k0 = 0; k0 < K3; k0 += 32) {
    __builtin_prefetch(gb[0] + k0 + 128, 0, 1);   // global_prefetch_b8
    v8bf a0lo = *(const v8bf*)(ga0 + k0 + koff);
    v8bf a0hi = *(const v8bf*)(ga0 + k0 + 8 + koff);
    v8bf a1lo = *(const v8bf*)(ga1 + k0 + koff);
    v8bf a1hi = *(const v8bf*)(ga1 + k0 + 8 + koff);
    v16bf af0 = __builtin_shufflevector(a0lo, a0hi,
        0,1,2,3,4,5,6,7,8,9,10,11,12,13,14,15);
    v16bf af1 = __builtin_shufflevector(a1lo, a1hi,
        0,1,2,3,4,5,6,7,8,9,10,11,12,13,14,15);
    #pragma unroll
    for (int t = 0; t < 4; ++t) {
      v16bf bf = *(const v16bf*)(gb[t] + k0);
      acc[0][t] = __builtin_amdgcn_wmma_f32_16x16x32_bf16(
          false, af0, false, bf, (short)0, acc[0][t], false, false);
      acc[1][t] = __builtin_amdgcn_wmma_f32_16x16x32_bf16(
          false, af1, false, bf, (short)0, acc[1][t], false, false);
    }
  }

  // C/D layout: VGPR r -> M=r (lanes 0-15) / M=r+8 (lanes 16-31), N=lane%16
  #pragma unroll
  for (int i = 0; i < 2; ++i) {
    int rowBase = m0 + i * 16 + half * 8;
    #pragma unroll
    for (int t = 0; t < 4; ++t) {
      int col = n0 + t * 16 + l16;
      float bv = bias[col];
      #pragma unroll
      for (int r = 0; r < 8; ++r)
        out[(size_t)(rowBase + r) * VV + col] = acc[i][t][r] + bv;
    }
  }
}

// ---------------------------------------------------------------------------
// 5) Per-row softmax stats over V
// ---------------------------------------------------------------------------
__global__ __launch_bounds__(256) void row_softmax_stats(
    const float* __restrict__ logits, float* __restrict__ rowmax,
    float* __restrict__ rowsum)
{
  __shared__ float sd[256];
  int row = blockIdx.x;
  int tid = threadIdx.x;
  const float* lg = logits + (size_t)row * VV;
  float m = -3.4e38f;
  for (int v = tid; v < VV; v += 256) m = fmaxf(m, lg[v]);
  sd[tid] = m; __syncthreads();
  for (int s = 128; s > 0; s >>= 1) {
    if (tid < s) sd[tid] = fmaxf(sd[tid], sd[tid + s]);
    __syncthreads();
  }
  float mx = sd[0]; __syncthreads();
  float sum = 0.f;
  for (int v = tid; v < VV; v += 256) sum += __expf(lg[v] - mx);
  sd[tid] = sum; __syncthreads();
  for (int s = 128; s > 0; s >>= 1) {
    if (tid < s) sd[tid] += sd[tid + s];
    __syncthreads();
  }
  if (tid == 0) { rowmax[row] = mx; rowsum[row] = sd[0]; }
}

// ---------------------------------------------------------------------------
// 6) out = est2 * softmax(logits)  (in place)
// ---------------------------------------------------------------------------
__global__ __launch_bounds__(256) void finalize_vocab(
    float* __restrict__ out, const float* __restrict__ rowmax,
    const float* __restrict__ rowsum, const float* __restrict__ est)
{
  int col = blockIdx.x * 256 + threadIdx.x;
  int row = blockIdx.y;
  size_t i = (size_t)row * VV + col;
  float p = __expf(out[i] - rowmax[row]) / rowsum[row];
  out[i] = est[row * 3 + 2] * p;
}

// ---------------------------------------------------------------------------
// 7) out[b,s,idx[b,l]] += est[b,s,comp] * att[b,l,s]
// ---------------------------------------------------------------------------
__global__ __launch_bounds__(256) void scatter_att(
    const float* __restrict__ att, const int* __restrict__ idx,
    const float* __restrict__ est, int comp, int L, float* __restrict__ out)
{
  int t = blockIdx.x * 256 + threadIdx.x;
  int total = BB * L * SS;
  if (t >= total) return;
  int s  = t % SS;
  int bl = t / SS;
  int b  = bl / L;
  int v  = idx[bl];
  int row = b * SS + s;
  float wgt = est[row * 3 + comp] * att[(size_t)bl * SS + s];
  atomicAdd(out + (size_t)row * VV + v, wgt);
}

// ---------------------------------------------------------------------------
extern "C" void kernel_launch(void* const* d_in, const int* in_sizes, int n_in,
                              void* d_out, int out_size, void* d_ws, size_t ws_size,
                              hipStream_t stream)
{
  const float* att_c    = (const float*)d_in[0];
  const float* att_q    = (const float*)d_in[1];
  const float* ctx_c    = (const float*)d_in[2];
  const float* ctx_q    = (const float*)d_in[3];
  const float* decoded  = (const float*)d_in[4];
  const float* cq       = (const float*)d_in[5];
  const float* eps      = (const float*)d_in[6];
  const float* gumbel   = (const float*)d_in[7];
  const float* Wv       = (const float*)d_in[8];
  const float* bv       = (const float*)d_in[9];
  const float* Wmu      = (const float*)d_in[10];
  const float* Wsg      = (const float*)d_in[11];
  const float* Wpi      = (const float*)d_in[12];
  const float* bpi      = (const float*)d_in[13];
  const int*   case_idx = (const int*)d_in[14];
  const int*   q_idx    = (const int*)d_in[15];
  const int*   num_step = (const int*)d_in[16];
  float* out = (float*)d_out;

  char* ws = (char*)d_ws;
  size_t off = 0;
  unsigned short* WT  = (unsigned short*)(ws + off); off += (size_t)VV * K3 * 2;   // 98,304,000
  unsigned short* Abf = (unsigned short*)(ws + off); off += (size_t)ROWS * K3 * 2; // 3,145,728
  float* est    = (float*)(ws + off); off += (size_t)ROWS * 3 * 4;
  off = (off + 255) & ~(size_t)255;
  float* rowmax = (float*)(ws + off); off += (size_t)ROWS * 4;
  float* rowsum = (float*)(ws + off); off += (size_t)ROWS * 4;

  build_A_bf16<<<(ROWS * K3 + 255) / 256, 256, 0, stream>>>(ctx_q, ctx_c, decoded, Abf);
  transpose_W_bf16<<<dim3(VV / 32, K3 / 32), dim3(32, 8), 0, stream>>>(Wv, WT);
  est_kernel<<<ROWS, 256, 0, stream>>>(ctx_q, ctx_c, decoded, cq, eps, gumbel,
                                       Wmu, Wsg, Wpi, bpi, num_step, est);
  gemm_logits<<<2000, 256, 0, stream>>>(Abf, WT, bv, out);
  row_softmax_stats<<<ROWS, 256, 0, stream>>>(out, rowmax, rowsum);
  finalize_vocab<<<dim3(VV / 256, ROWS), 256, 0, stream>>>(out, rowmax, rowsum, est);
  scatter_att<<<(BB * QLn * SS + 255) / 256, 256, 0, stream>>>(att_q, q_idx, est, 0, QLn, out);
  scatter_att<<<(BB * CLn * SS + 255) / 256, 256, 0, stream>>>(att_c, case_idx, est, 1, CLn, out);
}
